// RTDETRPostProcessor_43645457662111
// MI455X (gfx1250) — matrix operand order repaired and checked by reading, not compiled, not run on hardware
//
#include <hip/hip_runtime.h>
#include <hip/hip_bf16.h>
#include <math.h>

#define B_   2
#define Q_   300
#define C_   80
#define K_   300
#define MH_  128
#define MW_  128
#define TH_  320
#define TW_  320
#define QC_  (Q_*C_)              /* 24000 */
#define NCH  ((QC_ + 1023)/1024)  /* 24 chunks per thread */
#define STRIP_W 80
#define NSTRIP  4
#define HS_STR  (MH_ + 2)         /* padded LDS row stride (halves) */

typedef __attribute__((ext_vector_type(16))) _Float16 v16h;
typedef __attribute__((ext_vector_type(8)))  float    v8f;

// ---------------------------------------------------------------------------
// Kernel 1: per-batch top-300 over sigmoid(logits) flattened (q*c), plus box
// gather/convert. Incremental argmax: wave32 shuffle reduce + 32-entry LDS
// cross-wave reduce; only the owner thread of the removed element rescans.
// ---------------------------------------------------------------------------
__global__ __launch_bounds__(1024) void rtdetr_topk_kernel(
    const float* __restrict__ logits,   // [B,Q,C]
    const float* __restrict__ boxes,    // [B,Q,4] cxcywh in [0,1]
    const float* __restrict__ sizes,    // [B,2]   (w,h)
    float* __restrict__ out_scores,     // [B,K]
    float* __restrict__ out_labels,     // [B,K] (labels as float)
    float* __restrict__ out_boxes,      // [B,K,4]
    int*   __restrict__ q_ws)           // [B*K] selected query indices
{
    __shared__ float s_val[32];
    __shared__ int   s_idx[32];
    __shared__ int   s_gidx;

    const int b    = blockIdx.x;
    const int tid  = threadIdx.x;
    const int lane = tid & 31;
    const int wav  = tid >> 5;          // 0..31
    const float* lg = logits + b * QC_;

    unsigned rm = 0u;                   // private removal bitmask (24 bits)
    float v_loc = -3.0e38f;
    int   id_loc = 0x7FFFFFFF;
    #pragma unroll
    for (int j = 0; j < NCH; ++j) {
        int i = tid + j * 1024;
        if (i < QC_) {
            float x = lg[i];
            if (x > v_loc) { v_loc = x; id_loc = i; }
        }
    }

    for (int it = 0; it < K_; ++it) {
        float v = v_loc; int id = id_loc;
        #pragma unroll
        for (int off = 16; off > 0; off >>= 1) {
            float ov = __shfl_xor(v,  off, 32);
            int   oi = __shfl_xor(id, off, 32);
            if (ov > v || (ov == v && (unsigned)oi < (unsigned)id)) { v = ov; id = oi; }
        }
        if (lane == 0) { s_val[wav] = v; s_idx[wav] = id; }
        __syncthreads();
        if (wav == 0) {
            v = s_val[lane]; id = s_idx[lane];
            #pragma unroll
            for (int off = 16; off > 0; off >>= 1) {
                float ov = __shfl_xor(v,  off, 32);
                int   oi = __shfl_xor(id, off, 32);
                if (ov > v || (ov == v && (unsigned)oi < (unsigned)id)) { v = ov; id = oi; }
            }
            if (lane == 0) {
                s_gidx = id;
                float sc = 1.0f / (1.0f + __expf(-v));     // sigmoid of winner
                int   lab = id % C_;
                int   q   = id / C_;
                out_scores[b * K_ + it] = sc;
                out_labels[b * K_ + it] = (float)lab;
                q_ws[b * K_ + it] = q;
                const float* pb = boxes + (b * Q_ + q) * 4;
                float cx = pb[0], cy = pb[1], ww = pb[2], hh = pb[3];
                float W = sizes[b * 2 + 0], H = sizes[b * 2 + 1];
                float* ob = out_boxes + (b * K_ + it) * 4;
                ob[0] = (cx - 0.5f * ww) * W;
                ob[1] = (cy - 0.5f * hh) * H;
                ob[2] = (cx + 0.5f * ww) * W;
                ob[3] = (cy + 0.5f * hh) * H;
            }
        }
        __syncthreads();
        const int gid = s_gidx;
        if (tid == (gid & 1023)) {          // owner removes + rescans its chunk
            rm |= 1u << (gid >> 10);
            v_loc = -3.0e38f; id_loc = 0x7FFFFFFF;
            #pragma unroll
            for (int j = 0; j < NCH; ++j) {
                if (rm & (1u << j)) continue;
                int i = tid + j * 1024;
                if (i < QC_) {
                    float x = lg[i];
                    if (x > v_loc) { v_loc = x; id_loc = i; }
                }
            }
        }
        __syncthreads();
    }
}

// bilinear weight for source index k given (lo,hi,frac); reproduces
// half-pixel-center edge clamping exactly (lo==hi at edges -> weight 1).
__device__ __forceinline__ float interp_w(int k, int lo, int hi, float f) {
    float w = 0.0f;
    if (k == lo) w += 1.0f - f;
    if (k == hi) w += f;
    return w;
}

// ---------------------------------------------------------------------------
// Kernel 2: gather + bilinear resize 128x128 -> 320x320 via two separable
// f16 WMMA GEMM passes, then sign-threshold (sigmoid(x)>0.5 <=> x>0).
// Grid: (4 column strips of 80 out-cols) x (B*K masks). Block: 256 = 8 waves.
// The interpolation operators are 2-banded, so K is windowed:
//   Pass 1: H(128x80) = M(:,w1:w1+64) x Wcol(w1:w1+64, strip)   (K=64)
//   Pass 2: OUT tile  = Rh(rows, w2:w2+32) x H(w2:w2+32, :)     (K=32)
// H is stored transposed in padded LDS (f16) for contiguous-K, bank-spread
// reads. A fragments are hoisted and reused across all 5 column tiles.
// ---------------------------------------------------------------------------
__global__ __launch_bounds__(256) void rtdetr_mask_kernel(
    const float* __restrict__ masks,    // [B,Q,128,128]
    const int*   __restrict__ q_ws,     // [B*K]
    float* __restrict__ out)            // [B,K,320,320]
{
    __shared__ _Float16 Hs[STRIP_W * HS_STR];   // transposed: [oc_local][in_row]

    const int strip = blockIdx.x;            // 0..3
    const int mk    = blockIdx.y;            // 0..599  (b*K + k)
    const int b     = mk / K_;
    const int q     = q_ws[mk];
    const float* M  = masks + ((size_t)(b * Q_ + q)) * (MH_ * MW_);

    const int lane = threadIdx.x & 31;
    const int wav  = threadIdx.x >> 5;       // 0..7
    const int ln   = lane & 15;
    const int g    = lane >> 4;
    const int sc0  = strip * STRIP_W;
    const float scale = (float)MW_ / (float)TW_;   // 0.4

    // strip-uniform 64-wide input-column window (covers [32s-1, 32s+32])
    const int w1 = min(max(32 * strip - 16, 0), MW_ - 64);

    // ---- Pass 1: column interpolation (K = 64, two 32-chunks) ----
    {
        const int rt = wav;                  // 8 input-row tiles of 16
        const int m  = rt * 16 + ln;         // A lane row (mask row)
        const float* Mrow = M + m * MW_ + w1;

        // hoisted A fragments for both K-chunks (reused across 5 ct tiles)
        v16h a0, a1;
        #pragma unroll
        for (int v = 0; v < 8; ++v) {
            int p0 = (v < 4) ? (2 * v + 8 * g) : (16 + 2 * (v - 4) + 8 * g);
            float2 x0 = *(const float2*)(Mrow + p0);
            float2 x1 = *(const float2*)(Mrow + 32 + p0);
            a0[2 * v] = (_Float16)x0.x; a0[2 * v + 1] = (_Float16)x0.y;
            a1[2 * v] = (_Float16)x1.x; a1[2 * v + 1] = (_Float16)x1.y;
        }

        for (int ct = 0; ct < 5; ++ct) {
            const int oc = sc0 + ct * 16 + ln;       // B lane col (out col)
            float sx = ((float)oc + 0.5f) * scale - 0.5f;
            int   i0 = (int)floorf(sx);
            float f  = sx - (float)i0;
            int   lo = min(max(i0, 0), MW_ - 1);
            int   hi = min(max(i0 + 1, 0), MW_ - 1);

            v16h b0, b1;                             // weight fragments
            #pragma unroll
            for (int v = 0; v < 8; ++v) {
                int k0 = w1 + ((v < 4) ? (2 * v + 8 * g) : (16 + 2 * (v - 4) + 8 * g));
                b0[2 * v]     = (_Float16)interp_w(k0,      lo, hi, f);
                b0[2 * v + 1] = (_Float16)interp_w(k0 + 1,  lo, hi, f);
                b1[2 * v]     = (_Float16)interp_w(k0 + 32, lo, hi, f);
                b1[2 * v + 1] = (_Float16)interp_w(k0 + 33, lo, hi, f);
            }
            v8f acc = {};
            acc = __builtin_amdgcn_wmma_f32_16x16x32_f16(
                      false, a0, false, b0, (short)0, acc, false, false);
            acc = __builtin_amdgcn_wmma_f32_16x16x32_f16(
                      false, a1, false, b1, (short)0, acc, false, false);
            #pragma unroll
            for (int v = 0; v < 8; ++v) {            // D: lane=N col, vgpr=row
                int m_glob = rt * 16 + v + 8 * g;
                Hs[(ct * 16 + ln) * HS_STR + m_glob] = (_Float16)acc[v];
            }
        }
    }
    __syncthreads();

    // ---- Pass 2: row interpolation + binarize (K = 32 per row tile) ----
    float* O = out + (size_t)mk * (TH_ * TW_);
    for (int rt = wav; rt < 20; rt += 8) {           // 20 out-row tiles
        // tile-uniform 32-wide input-row window (row support span <= 8)
        int   i0min = (int)floorf(((float)(rt * 16) + 0.5f) *
                                  ((float)MH_ / (float)TH_) - 0.5f);
        const int w2 = min(max(i0min - 8, 0), MH_ - 32);

        const int orow = rt * 16 + ln;               // A lane row (out row)
        float sy = ((float)orow + 0.5f) * ((float)MH_ / (float)TH_) - 0.5f;
        int   i0 = (int)floorf(sy);
        float f  = sy - (float)i0;
        int   lo = min(max(i0, 0), MH_ - 1);
        int   hi = min(max(i0 + 1, 0), MH_ - 1);

        // hoisted Rh fragment (reused across 5 column tiles)
        v16h a;
        #pragma unroll
        for (int v = 0; v < 8; ++v) {
            int k0 = w2 + ((v < 4) ? (2 * v + 8 * g) : (16 + 2 * (v - 4) + 8 * g));
            a[2 * v]     = (_Float16)interp_w(k0,     lo, hi, f);
            a[2 * v + 1] = (_Float16)interp_w(k0 + 1, lo, hi, f);
        }

        for (int ct = 0; ct < 5; ++ct) {
            const int ocl = ct * 16 + ln;            // B lane col (strip-local)
            v16h bb;                                 // H fragment from LDS
            #pragma unroll
            for (int v = 0; v < 8; ++v) {
                int k0 = w2 + ((v < 4) ? (2 * v + 8 * g) : (16 + 2 * (v - 4) + 8 * g));        // table-order K pair
                bb[2 * v]     = Hs[ocl * HS_STR + k0];
                bb[2 * v + 1] = Hs[ocl * HS_STR + k0 + 1];
            }
            v8f acc = {};
            acc = __builtin_amdgcn_wmma_f32_16x16x32_f16(
                      false, a, false, bb, (short)0, acc, false, false);
            #pragma unroll
            for (int v = 0; v < 8; ++v) {
                int orr = rt * 16 + v + 8 * g;
                int occ = sc0 + ct * 16 + ln;
                O[orr * TW_ + occ] = (acc[v] > 0.0f) ? 1.0f : 0.0f;
            }
        }
    }
}

extern "C" void kernel_launch(void* const* d_in, const int* in_sizes, int n_in,
                              void* d_out, int out_size, void* d_ws, size_t ws_size,
                              hipStream_t stream) {
    (void)in_sizes; (void)n_in; (void)out_size; (void)ws_size;

    const float* logits = (const float*)d_in[0];   // [B,Q,C]
    const float* boxes  = (const float*)d_in[1];   // [B,Q,4]
    const float* masks  = (const float*)d_in[2];   // [B,Q,128,128]
    const float* sizes  = (const float*)d_in[3];   // [B,2]
    // d_in[4], d_in[5] = target_h/target_w (compile-time constants here)

    float* out        = (float*)d_out;
    float* out_scores = out;                               // 600
    float* out_labels = out + B_ * K_;                     // 600
    float* out_boxes  = out + 2 * B_ * K_;                 // 2400
    float* out_masks  = out + 2 * B_ * K_ + B_ * K_ * 4;   // 61,440,000
    int*   q_ws       = (int*)d_ws;                        // [B*K]

    rtdetr_topk_kernel<<<B_, 1024, 0, stream>>>(
        logits, boxes, sizes, out_scores, out_labels, out_boxes, q_ws);

    dim3 grid(NSTRIP, B_ * K_);
    rtdetr_mask_kernel<<<grid, 256, 0, stream>>>(masks, q_ws, out_masks);
}
